// MambaBlock_26946624815373
// MI455X (gfx1250) — matrix operand order repaired and checked by reading, not compiled, not run on hardware
//
#include <hip/hip_runtime.h>
#include <hip/hip_bf16.h>

// ---------------------------------------------------------------------------
// Mamba block for MI455X (gfx1250, wave32, WMMA bf16 -> f32 accumulate,
// async global->LDS double-buffered GEMM pipeline)
// ---------------------------------------------------------------------------

#define D_MODEL   1024
#define D_STATE   16
#define D_CONV    4
#define D_INNER   2048
#define DT_RANK   64
#define NPROJ     (DT_RANK + 2 * D_STATE)   // 96
#define B_SZ      4
#define SEQ_L     2048
#define BL        (B_SZ * SEQ_L)            // 8192 rows

typedef __attribute__((ext_vector_type(8)))  unsigned short v8u16;
typedef __attribute__((ext_vector_type(16))) __bf16         v16bf;
typedef __attribute__((ext_vector_type(8)))  float          v8f;
typedef int v4i_b128 __attribute__((vector_size(16)));   // matches builtin param

union FragAB { v8u16 h[2]; v16bf v; };

#if __has_builtin(__builtin_amdgcn_global_load_async_to_lds_b128)
#define HAVE_ASYNC_LDS 1
#endif
#if __has_builtin(__builtin_amdgcn_s_wait_asynccnt)
#define HAVE_WAIT_ASYNC 1
#endif

// async 16-byte global -> LDS copy (ASYNCcnt-tracked)
__device__ __forceinline__ void async_cp16(const void* g, void* l) {
#if defined(HAVE_ASYNC_LDS)
  __builtin_amdgcn_global_load_async_to_lds_b128(
      (__attribute__((address_space(1))) v4i_b128*)(void*)g,
      (__attribute__((address_space(3))) v4i_b128*)l, 0, 0);
#else
  unsigned int loff =
      (unsigned int)(unsigned long long)(__attribute__((address_space(3))) void*)l;
  asm volatile("global_load_async_to_lds_b128 %0, %1, off"
               :: "v"(loff), "v"(g) : "memory");
#endif
}

__device__ __forceinline__ void async_wait0() {
#if defined(HAVE_WAIT_ASYNC)
  __builtin_amdgcn_s_wait_asynccnt(0);
#else
  asm volatile("s_wait_asynccnt 0x0" ::: "memory");
#endif
}

// round-to-nearest-even f32 -> bf16 (as raw u16)
__device__ __forceinline__ unsigned short f2bf(float f) {
  unsigned int u = __float_as_uint(f);
  unsigned int r = u + 0x7FFFu + ((u >> 16) & 1u);
  return (unsigned short)(r >> 16);
}

// ---------------------------------------------------------------------------
// elementwise f32 -> bf16
// ---------------------------------------------------------------------------
__global__ void cvt_f32_bf16(const float* __restrict__ src,
                             unsigned short* __restrict__ dst, long n) {
  long i = (long)blockIdx.x * blockDim.x + threadIdx.x;
  if (i < n) dst[i] = f2bf(src[i]);
}

// slice x_dbl[:, 0:64] -> bf16 (rows x 96 f32 -> rows x 64 bf16)
__global__ void cvt_dt_slice(const float* __restrict__ xdbl,
                             unsigned short* __restrict__ dt, long rows) {
  long i = (long)blockIdx.x * blockDim.x + threadIdx.x;
  if (i < rows * DT_RANK) {
    long r = i >> 6;
    int  j = (int)(i & 63);
    dt[i] = f2bf(xdbl[r * NPROJ + j]);
  }
}

// ---------------------------------------------------------------------------
// LayerNorm (eps=1e-5) over D_MODEL, output bf16
// ---------------------------------------------------------------------------
__global__ __launch_bounds__(256)
void layernorm_bf16(const float* __restrict__ x, const float* __restrict__ w,
                    const float* __restrict__ b, unsigned short* __restrict__ o) {
  __shared__ float rs[256], rs2[256];
  const long row = blockIdx.x;
  const float* xr = x + row * (long)D_MODEL;
  float s = 0.f, s2 = 0.f;
  for (int i = threadIdx.x; i < D_MODEL; i += 256) {
    float v = xr[i]; s += v; s2 += v * v;
  }
  rs[threadIdx.x] = s; rs2[threadIdx.x] = s2;
  __syncthreads();
  for (int off = 128; off > 0; off >>= 1) {
    if ((int)threadIdx.x < off) {
      rs[threadIdx.x]  += rs[threadIdx.x + off];
      rs2[threadIdx.x] += rs2[threadIdx.x + off];
    }
    __syncthreads();
  }
  const float mu   = rs[0] * (1.f / D_MODEL);
  const float var  = rs2[0] * (1.f / D_MODEL) - mu * mu;
  const float rstd = rsqrtf(var + 1e-5f);
  for (int i = threadIdx.x; i < D_MODEL; i += 256)
    o[row * D_MODEL + i] = f2bf((xr[i] - mu) * rstd * w[i] + b[i]);
}

// ---------------------------------------------------------------------------
// WMMA GEMM:  C[M,N] (f32) = A[M,K] (bf16, row-major) x W[N,K]^T (bf16)
//
// 256-thread block = 8 waves computing a 128x64 tile (wave = 64x16 micro-tile
// of 4 stacked 16x16 WMMAs).  Per 32-K step the block stages A(128x32) and
// W(64x32) into LDS with global_load_async_to_lds_b128 (double buffered,
// s_wait_asynccnt + barrier), then waves read fragments from LDS.
//
// LDS rows are padded to 40 shorts (80B) so 16-lane b128 reads are
// bank-conflict-free (gcd(20 dwords, 64 banks) cycle covers 16 lanes).
//
// Fragment packing used here (contiguous k order, same for A and B):
//   A 16x32:  lane = half*16 + row;  elems j -> K=k0+16*half+j
//   B 32x16:  lane = half*16 + col;  elems j -> K=k0+16*half+j  (B[k][n]=W[n*K+k])
//   D 16x16:  lane = half*16 + col;  vgpr r -> row m = 8*half + r
// ---------------------------------------------------------------------------
enum { EPI_NONE = 0, EPI_SOFTPLUS = 1, EPI_RESID = 2 };

#define KC 32      // K-step
#define MT 128     // block M tile
#define NT 64      // block N tile
#define AP 40      // padded LDS row stride (shorts)

template <int EPI>
__global__ __launch_bounds__(256)
void gemm_bf16_wmma(const unsigned short* __restrict__ A,
                    const unsigned short* __restrict__ W,
                    float* __restrict__ C,
                    const float* __restrict__ bias,
                    const float* __restrict__ res,
                    int M, int N, int K) {
  __shared__ unsigned short sA[2][MT * AP];   // 2 x 10240 B
  __shared__ unsigned short sW[2][NT * AP];   // 2 x  5120 B

  const int tid  = threadIdx.x;
  const int lane = tid & 31;
  const int wave = tid >> 5;        // 0..7
  const int mgrp = wave >> 2;       // 0..1 -> rows tm + mgrp*64
  const int ngrp = wave & 3;        // 0..3 -> cols tn + ngrp*16
  const int half = lane >> 4;
  const int lr   = lane & 15;

  const long tm = (long)blockIdx.x * MT;
  const long tn = (long)blockIdx.y * NT;

  // copy-chunk mapping: 16B chunks, 4 per row of 32 shorts
  const int crow = tid >> 2;        // 0..63
  const int cq   = tid & 3;         // 0..3

  v8f c[4];
#pragma unroll
  for (int i = 0; i < 4; ++i)
#pragma unroll
    for (int r = 0; r < 8; ++r) c[i][r] = 0.f;

  auto issue = [&](int buf, int k0) {
    // A tile: 128 rows -> two chunks per thread
    async_cp16(A + (tm + crow) * (long)K + k0 + cq * 8,
               &sA[buf][crow * AP + cq * 8]);
    async_cp16(A + (tm + crow + 64) * (long)K + k0 + cq * 8,
               &sA[buf][(crow + 64) * AP + cq * 8]);
    // W tile: 64 rows -> one chunk per thread (clamp for partial N tile)
    long wr = tn + crow; if (wr >= N) wr = N - 1;
    async_cp16(W + wr * (long)K + k0 + cq * 8,
               &sW[buf][crow * AP + cq * 8]);
  };

  const int KT = K / KC;
  issue(0, 0);
  for (int kt = 0; kt < KT; ++kt) {
    const int cur = kt & 1;
    async_wait0();                      // this wave's tile-fill writes landed
    __syncthreads();                    // all waves' fills landed
    if (kt + 1 < KT) issue(cur ^ 1, (kt + 1) * KC);

    FragAB bf;
    const unsigned short* wb = &sW[cur][(ngrp * 16 + lr) * AP + half * 16];
    bf.h[0] = *(const v8u16*)(wb);
    bf.h[1] = *(const v8u16*)(wb + 8);
#pragma unroll
    for (int i = 0; i < 4; ++i) {
      const unsigned short* ab =
          &sA[cur][(mgrp * 64 + i * 16 + lr) * AP + half * 16];
      FragAB af;
      af.h[0] = *(const v8u16*)(ab);
      af.h[1] = *(const v8u16*)(ab + 8);
      c[i] = __builtin_amdgcn_wmma_f32_16x16x32_bf16(
          /*neg_a=*/false, af.v, /*neg_b=*/false, bf.v,
          /*c_mod=*/(short)0, c[i], /*reuse_a=*/false, /*reuse_b=*/false);
    }
    __syncthreads();                    // done reading 'cur' before refill
  }

  const long ncol = tn + ngrp * 16 + lr;
  if ((long)(tn + ngrp * 16) < (long)N) {
#pragma unroll
    for (int i = 0; i < 4; ++i) {
#pragma unroll
      for (int r = 0; r < 8; ++r) {
        const long m   = tm + mgrp * 64 + i * 16 + half * 8 + r;
        const long idx = m * (long)N + ncol;
        float acc = c[i][r];
        if constexpr (EPI == EPI_SOFTPLUS) {
          float v = acc + bias[ncol];
          C[idx] = (v > 20.f) ? v : log1pf(__expf(v));
        } else if constexpr (EPI == EPI_RESID) {
          C[idx] = res[idx] + acc;
        } else {
          C[idx] = acc;
        }
      }
    }
  }
}

// ---------------------------------------------------------------------------
// causal depthwise conv1d (width 4) + SiLU; reads xi = xz[:, 0:D_INNER]
// writes u (f32) and u (bf16, GEMM operand)
// ---------------------------------------------------------------------------
__global__ __launch_bounds__(256)
void conv_silu(const float* __restrict__ xz, const float* __restrict__ cw,
               const float* __restrict__ cb, float* __restrict__ u,
               unsigned short* __restrict__ ubf) {
  long idx = (long)blockIdx.x * blockDim.x + threadIdx.x;   // over BL*D_INNER
  if (idx >= (long)BL * D_INNER) return;
  const int  d  = (int)(idx & (D_INNER - 1));
  const long bl = idx >> 11;                                 // row in [0, BL)
  const int  l  = (int)(bl & (SEQ_L - 1));
  float acc = cb[d];
#pragma unroll
  for (int j = 0; j < D_CONV; ++j) {
    int back = (D_CONV - 1) - j;                             // l - 3 + j
    if (l - back >= 0)
      acc += cw[d * D_CONV + j] * xz[(bl - back) * (long)(2 * D_INNER) + d];
  }
  const float uu = acc / (1.f + __expf(-acc));               // silu
  u[idx]   = uu;
  ubf[idx] = f2bf(uu);
}

// ---------------------------------------------------------------------------
// selective scan: one thread per (b, d) channel, 16 states in registers.
// epilogue fused: y = ys + u*D;  out = y * silu(z);  stores bf16 for out_proj.
// ---------------------------------------------------------------------------
__global__ __launch_bounds__(256)
void selective_scan(const float* __restrict__ delta, const float* __restrict__ u,
                    const float* __restrict__ xdbl, const float* __restrict__ xz,
                    const float* __restrict__ A_log, const float* __restrict__ Dp,
                    unsigned short* __restrict__ oact) {
  const int t = blockIdx.x * blockDim.x + threadIdx.x;
  if (t >= B_SZ * D_INNER) return;
  const int b = t >> 11;
  const int d = t & (D_INNER - 1);

  float Ad[D_STATE];
#pragma unroll
  for (int n = 0; n < D_STATE; ++n) Ad[n] = -__expf(A_log[d * D_STATE + n]);
  float h[D_STATE];
#pragma unroll
  for (int n = 0; n < D_STATE; ++n) h[n] = 0.f;
  const float Dd = Dp[d];

  for (int l = 0; l < SEQ_L; ++l) {
    const long row = (long)b * SEQ_L + l;
    const float dl = delta[row * D_INNER + d];
    const float uu = u[row * D_INNER + d];
    const float du = dl * uu;
    const float* Br = xdbl + row * NPROJ + DT_RANK;
    const float* Cr = Br + D_STATE;
    float y = 0.f;
#pragma unroll
    for (int n = 0; n < D_STATE; ++n) {
      h[n] = h[n] * __expf(dl * Ad[n]) + du * Br[n];
      y += h[n] * Cr[n];
    }
    const float yf = y + uu * Dd;
    const float z  = xz[row * (long)(2 * D_INNER) + D_INNER + d];
    const float g  = yf * (z / (1.f + __expf(-z)));
    oact[row * D_INNER + d] = f2bf(g);
  }
}

// ---------------------------------------------------------------------------
// launch
// ---------------------------------------------------------------------------
static inline size_t alignup(size_t x) { return (x + 255) & ~(size_t)255; }

extern "C" void kernel_launch(void* const* d_in, const int* in_sizes, int n_in,
                              void* d_out, int out_size, void* d_ws, size_t ws_size,
                              hipStream_t stream) {
  (void)in_sizes; (void)n_in; (void)out_size; (void)ws_size;

  const float* x         = (const float*)d_in[0];
  const float* norm_w    = (const float*)d_in[1];
  const float* norm_b    = (const float*)d_in[2];
  const float* in_proj_w = (const float*)d_in[3];   // (4096, 1024)
  const float* conv_w    = (const float*)d_in[4];   // (2048, 1, 4)
  const float* conv_b    = (const float*)d_in[5];
  const float* x_proj_w  = (const float*)d_in[6];   // (96, 2048)
  const float* dt_proj_w = (const float*)d_in[7];   // (2048, 64)
  const float* dt_proj_b = (const float*)d_in[8];
  const float* A_log     = (const float*)d_in[9];   // (2048, 16)
  const float* Dp        = (const float*)d_in[10];
  const float* out_proj_w= (const float*)d_in[11];  // (1024, 2048)
  float* out = (float*)d_out;                       // (BL, 1024)

  char* ws = (char*)d_ws;
  size_t off = 0;
  auto take = [&](size_t bytes) { char* p = ws + off; off = alignup(off + bytes); return p; };

  unsigned short* xn_bf   = (unsigned short*)take((size_t)BL * D_MODEL * 2);
  unsigned short* w_in_bf = (unsigned short*)take((size_t)2 * D_INNER * D_MODEL * 2);
  unsigned short* w_x_bf  = (unsigned short*)take((size_t)NPROJ * D_INNER * 2);
  unsigned short* w_dt_bf = (unsigned short*)take((size_t)D_INNER * DT_RANK * 2);
  unsigned short* w_out_bf= (unsigned short*)take((size_t)D_MODEL * D_INNER * 2);
  float*          xz      = (float*)take((size_t)BL * 2 * D_INNER * 4);
  float*          u_f     = (float*)take((size_t)BL * D_INNER * 4);
  unsigned short* u_bf    = (unsigned short*)take((size_t)BL * D_INNER * 2);
  float*          xdbl    = (float*)take((size_t)BL * NPROJ * 4);
  unsigned short* dt_bf   = (unsigned short*)take((size_t)BL * DT_RANK * 2);
  float*          dlt     = (float*)take((size_t)BL * D_INNER * 4);
  unsigned short* oact_bf = (unsigned short*)take((size_t)BL * D_INNER * 2);

  const int T = 256;
  auto nb = [&](long n) { return (unsigned int)((n + T - 1) / T); };

  // weight conversions f32 -> bf16
  cvt_f32_bf16<<<nb((long)2 * D_INNER * D_MODEL), T, 0, stream>>>(in_proj_w, w_in_bf, (long)2 * D_INNER * D_MODEL);
  cvt_f32_bf16<<<nb((long)NPROJ * D_INNER), T, 0, stream>>>(x_proj_w, w_x_bf, (long)NPROJ * D_INNER);
  cvt_f32_bf16<<<nb((long)D_INNER * DT_RANK), T, 0, stream>>>(dt_proj_w, w_dt_bf, (long)D_INNER * DT_RANK);
  cvt_f32_bf16<<<nb((long)D_MODEL * D_INNER), T, 0, stream>>>(out_proj_w, w_out_bf, (long)D_MODEL * D_INNER);

  // 1) LayerNorm -> bf16
  layernorm_bf16<<<BL, 256, 0, stream>>>(x, norm_w, norm_b, xn_bf);

  // 2) in_proj: (8192x1024)x(1024x4096)
  gemm_bf16_wmma<EPI_NONE><<<dim3(BL / MT, (2 * D_INNER) / NT), 256, 0, stream>>>(
      xn_bf, w_in_bf, xz, nullptr, nullptr, BL, 2 * D_INNER, D_MODEL);

  // 3) depthwise conv + silu
  conv_silu<<<nb((long)BL * D_INNER), T, 0, stream>>>(xz, conv_w, conv_b, u_f, u_bf);

  // 4) x_proj: (8192x2048)x(2048x96)   (N=96 -> 2 partial N tiles)
  gemm_bf16_wmma<EPI_NONE><<<dim3(BL / MT, (NPROJ + NT - 1) / NT), 256, 0, stream>>>(
      u_bf, w_x_bf, xdbl, nullptr, nullptr, BL, NPROJ, D_INNER);

  // 5) dt slice -> bf16 ; dt_proj + softplus
  cvt_dt_slice<<<nb((long)BL * DT_RANK), T, 0, stream>>>(xdbl, dt_bf, BL);
  gemm_bf16_wmma<EPI_SOFTPLUS><<<dim3(BL / MT, D_INNER / NT), 256, 0, stream>>>(
      dt_bf, w_dt_bf, dlt, dt_proj_b, nullptr, BL, D_INNER, DT_RANK);

  // 6) selective scan (+ D skip + silu(z) gate, bf16 output)
  selective_scan<<<nb((long)B_SZ * D_INNER), T, 0, stream>>>(
      dlt, u_f, xdbl, xz, A_log, Dp, oact_bf);

  // 7) out_proj + residual: (8192x2048)x(2048x1024) + x
  gemm_bf16_wmma<EPI_RESID><<<dim3(BL / MT, D_MODEL / NT), 256, 0, stream>>>(
      oact_bf, w_out_bf, out, nullptr, x, BL, D_MODEL, D_INNER);
}
